// FlashMultiHeadAttention_43215960933244
// MI455X (gfx1250) — matrix operand order repaired and checked by reading, not compiled, not run on hardware
//
#include <hip/hip_runtime.h>
#include <hip/hip_bf16.h>

#define D_MODEL 1024
#define NUM_HEADS 16
#define HEAD_DIM 64
#define BATCH 4
#define SEQ 2048
#define MTOT (BATCH * SEQ)        // 8192 rows
#define BHTOT (BATCH * NUM_HEADS) // 64 (b,h) pairs

typedef __bf16 bf16_t;
typedef __attribute__((ext_vector_type(16))) __bf16 v16bf;
typedef __attribute__((ext_vector_type(8))) float v8f;

// ---------------------------------------------------------------------------
// D = A(16x32 bf16) * B(32x16 bf16) + C(16x16 f32), wave32 WMMA
// ---------------------------------------------------------------------------
static __device__ __forceinline__ v8f wmma_bf16(v16bf a, v16bf b, v8f c) {
  return __builtin_amdgcn_wmma_f32_16x16x32_bf16(
      /*neg_a=*/false, a, /*neg_b=*/false, b,
      /*c_mod=*/(short)0, c, /*reuse_a=*/false, /*reuse_b=*/false);
}

// LDS byte-offset of a shared-memory pointer (AS(3) pointers are 32-bit)
static __device__ __forceinline__ unsigned lds_off(const void* p) {
  return (unsigned)(uintptr_t)(__attribute__((address_space(3))) const void*)p;
}

// One 16B lane-chunk: global -> LDS via the async data path (ASYNCcnt)
static __device__ __forceinline__ void async_copy16(unsigned lds_byte,
                                                    const void* gaddr) {
  asm volatile("global_load_async_to_lds_b128 %0, %1, off"
               :: "v"(lds_byte), "v"(gaddr) : "memory");
}

static __device__ __forceinline__ void wait_async_zero() {
  asm volatile("s_wait_asynccnt 0x0" ::: "memory");
}

// A-fragment (16x32, row-major source, row stride ld):
// lane l: row m = l&15; element e -> K = (l>>4)*16 + e  (contiguous in k)
static __device__ __forceinline__ v16bf load_a(const bf16_t* __restrict__ src,
                                               int ld, int lane) {
  const int m = lane & 15, half = lane >> 4;
  const bf16_t* p = src + m * ld + half * 16;
  v16bf a;
#pragma unroll
  for (int e = 0; e < 16; ++e) a[e] = p[e];
  return a;
}

static __device__ __forceinline__ v16bf load_a(const float* __restrict__ src,
                                               int ld, int lane) {
  const int m = lane & 15, half = lane >> 4;
  const float* p = src + m * ld + half * 16;
  v16bf a;
#pragma unroll
  for (int e = 0; e < 16; ++e) a[e] = (bf16_t)p[e];
  return a;
}

// B-fragment where B[k][n] = src[n][k] (src row-major, stride ld):
// lane l: col n = l&15; element e -> K = (l>>4)*16 + e  (contiguous in k)
static __device__ __forceinline__ v16bf load_bt(const bf16_t* __restrict__ src,
                                                int ld, int lane) {
  const int n = lane & 15, half = lane >> 4;
  const bf16_t* p = src + n * ld + half * 16;
  v16bf b;
#pragma unroll
  for (int e = 0; e < 16; ++e) b[e] = p[e];
  return b;
}

static __device__ __forceinline__ v16bf load_bt(const float* __restrict__ src,
                                                int ld, int lane) {
  const int n = lane & 15, half = lane >> 4;
  const float* p = src + n * ld + half * 16;
  v16bf b;
#pragma unroll
  for (int e = 0; e < 16; ++e) b[e] = (bf16_t)p[e];
  return b;
}

// ---------------------------------------------------------------------------
// fp32 -> bf16 bulk conversion (one pass; removes cvt from GEMM hot loops)
// ---------------------------------------------------------------------------
__global__ __launch_bounds__(256) void cvt_f32_bf16_kernel(
    const float* __restrict__ in, bf16_t* __restrict__ out, int n4) {
  const int i = blockIdx.x * 256 + threadIdx.x;
  if (i < n4) {
    const float4 f = ((const float4*)in)[i];
    bf16_t* o = out + (size_t)i * 4;
    o[0] = (bf16_t)f.x; o[1] = (bf16_t)f.y;
    o[2] = (bf16_t)f.z; o[3] = (bf16_t)f.w;
  }
}

// ---------------------------------------------------------------------------
// Input projection: Y = X @ W^T + bias, scattered to head layout.
//   transpose_out == 0 : out[(bh*SEQ + s)*64 + d]        (Qh / Kh)
//   transpose_out == 1 : out[(bh*64 + d)*SEQ + s]        (Vt, for P@V B-frags)
// Wave tile 32x32 (2 A-frags x 2 B-frags -> 4 WMMA per K-step).
// grid = (MTOT/32, D_MODEL/128), block = 128 (4 waves)
// ---------------------------------------------------------------------------
template <typename TA, typename TW>
__global__ __launch_bounds__(128) void mha_proj_kernel(
    const TA* __restrict__ X, const TW* __restrict__ W,
    const float* __restrict__ bias, bf16_t* __restrict__ out,
    int transpose_out) {
  const int lane = threadIdx.x & 31;
  const int wave = threadIdx.x >> 5;
  const int m0 = blockIdx.x * 32;
  const int n0 = blockIdx.y * 128 + wave * 32;

  v8f c00 = {}, c01 = {}, c10 = {}, c11 = {};
#pragma unroll 2
  for (int k0 = 0; k0 < D_MODEL; k0 += 32) {
    const v16bf a0 = load_a(X + (long)m0 * D_MODEL + k0, D_MODEL, lane);
    const v16bf a1 = load_a(X + (long)(m0 + 16) * D_MODEL + k0, D_MODEL, lane);
    const v16bf b0 = load_bt(W + (long)n0 * D_MODEL + k0, D_MODEL, lane);
    const v16bf b1 = load_bt(W + (long)(n0 + 16) * D_MODEL + k0, D_MODEL, lane);
    c00 = wmma_bf16(a0, b0, c00);
    c01 = wmma_bf16(a0, b1, c01);
    c10 = wmma_bf16(a1, b0, c10);
    c11 = wmma_bf16(a1, b1, c11);
  }

  const int n = lane & 15, half = lane >> 4;
  const v8f accs[4] = {c00, c01, c10, c11};
#pragma unroll
  for (int t = 0; t < 4; ++t) {
    const int mbase = m0 + (t >> 1) * 16;
    const int col = n0 + (t & 1) * 16 + n;
    const float bn = bias[col];
    const int h = col >> 6;  // head index
    const int d = col & 63;  // dim within head
#pragma unroll
    for (int r = 0; r < 8; ++r) {
      const int m = mbase + r + 8 * half;
      const int b_idx = m >> 11;  // / SEQ
      const int s = m & (SEQ - 1);
      const float y = accs[t][r] + bn;
      const long bh = (long)b_idx * NUM_HEADS + h;
      if (transpose_out)
        out[(bh * HEAD_DIM + d) * SEQ + s] = (bf16_t)y;
      else
        out[(bh * SEQ + s) * HEAD_DIM + d] = (bf16_t)y;
    }
  }
}

// ---------------------------------------------------------------------------
// Output projection: out(f32) = A @ W^T + bias, plain [M, D] layout.
// ---------------------------------------------------------------------------
template <typename TA, typename TW>
__global__ __launch_bounds__(128) void mha_outproj_kernel(
    const TA* __restrict__ A, const TW* __restrict__ W,
    const float* __restrict__ bias, float* __restrict__ out) {
  const int lane = threadIdx.x & 31;
  const int wave = threadIdx.x >> 5;
  const int m0 = blockIdx.x * 32;
  const int n0 = blockIdx.y * 128 + wave * 32;

  v8f c00 = {}, c01 = {}, c10 = {}, c11 = {};
#pragma unroll 2
  for (int k0 = 0; k0 < D_MODEL; k0 += 32) {
    const v16bf a0 = load_a(A + (long)m0 * D_MODEL + k0, D_MODEL, lane);
    const v16bf a1 = load_a(A + (long)(m0 + 16) * D_MODEL + k0, D_MODEL, lane);
    const v16bf b0 = load_bt(W + (long)n0 * D_MODEL + k0, D_MODEL, lane);
    const v16bf b1 = load_bt(W + (long)(n0 + 16) * D_MODEL + k0, D_MODEL, lane);
    c00 = wmma_bf16(a0, b0, c00);
    c01 = wmma_bf16(a0, b1, c01);
    c10 = wmma_bf16(a1, b0, c10);
    c11 = wmma_bf16(a1, b1, c11);
  }

  const int n = lane & 15, half = lane >> 4;
  const v8f accs[4] = {c00, c01, c10, c11};
#pragma unroll
  for (int t = 0; t < 4; ++t) {
    const int mbase = m0 + (t >> 1) * 16;
    const int col = n0 + (t & 1) * 16 + n;
    const float bn = bias[col];
#pragma unroll
    for (int r = 0; r < 8; ++r) {
      const int m = mbase + r + 8 * half;
      out[(long)m * D_MODEL + col] = accs[t][r] + bn;
    }
  }
}

// ---------------------------------------------------------------------------
// Flash attention core. grid = (BHTOT, SEQ/64), block = 128 (4 waves).
// K/V blocks staged with GLOBAL_LOAD_ASYNC_TO_LDS_B128 (ASYNCcnt path);
// each wave owns a 16-row Q tile; online softmax with running (m, l).
// ---------------------------------------------------------------------------
__global__ __launch_bounds__(128) void mha_attn_kernel(
    const bf16_t* __restrict__ Qh, const bf16_t* __restrict__ Kh,
    const bf16_t* __restrict__ Vt, bf16_t* __restrict__ attn) {
  __shared__ bf16_t Ks[64 * 64];      // [key][hd]
  __shared__ bf16_t Vs[64 * 64];      // [hd][key]   (transposed V block)
  __shared__ bf16_t Ps[4 * 16 * 64];  // per-wave P scratch [row][key]

  const int lane = threadIdx.x & 31;
  const int wave = threadIdx.x >> 5;
  const int bh = blockIdx.x;
  const int q0 = blockIdx.y * 64 + wave * 16;

  // Q A-fragments are loop-invariant: 16 rows x 64 (two 16x32 fragments)
  const bf16_t* Qbase = Qh + ((long)bh * SEQ + q0) * HEAD_DIM;
  const v16bf aq0 = load_a(Qbase, HEAD_DIM, lane);
  const v16bf aq1 = load_a(Qbase + 32, HEAD_DIM, lane);

  v8f o0 = {}, o1 = {}, o2 = {}, o3 = {};
  float mrow[8], lrow[8];
#pragma unroll
  for (int r = 0; r < 8; ++r) { mrow[r] = -1e30f; lrow[r] = 0.0f; }

  const float scale = 0.125f;  // 1/sqrt(64)
  bf16_t* Pw = Ps + wave * 16 * 64;
  const int half = lane >> 4;
  const unsigned ks0 = lds_off(Ks);
  const int dd = threadIdx.x >> 1;   // V staging: hd row per thread pair
  const int seg = threadIdx.x & 1;
  const unsigned vs0 = lds_off(Vs) + (unsigned)dd * 128u;

  for (int kb = 0; kb < SEQ; kb += 64) {
    // ---- async-stage K block (contiguous 8KB) : 4 x 16B per thread ----
    {
      const char* kg = (const char*)(Kh + ((long)bh * SEQ + kb) * HEAD_DIM);
#pragma unroll
      for (int j = 0; j < 4; ++j) {
        const unsigned idx = (unsigned)threadIdx.x + 128u * j;
        async_copy16(ks0 + idx * 16u, kg + (size_t)idx * 16);
      }
    }
    // ---- async-stage V^T block: 64 hd-rows x 128B, rows strided by SEQ ----
    {
      const char* vg = (const char*)(Vt + ((long)bh * HEAD_DIM + dd) * SEQ + kb);
#pragma unroll
      for (int i = 0; i < 4; ++i) {
        const unsigned off = (unsigned)(seg * 4 + i) * 16u;
        async_copy16(vs0 + off, vg + off);
      }
    }
    wait_async_zero();   // ASYNCcnt -> 0 before the workgroup barrier
    __syncthreads();

    // prefetch next K block into the cache hierarchy (global_prefetch_b8)
    if (kb + 64 < SEQ) {
      const char* nk = (const char*)(Kh + ((long)bh * SEQ + kb + 64) * HEAD_DIM);
      __builtin_prefetch(nk + (size_t)threadIdx.x * 64, 0, 1);
    }

    // ---- S = Q @ K^T : four 16x16 n-tiles, two k-chunks each ----
    v8f sacc[4];
#pragma unroll
    for (int nt = 0; nt < 4; ++nt) {
      v8f c = {};
      c = wmma_bf16(aq0, load_bt(Ks + nt * 16 * 64, 64, lane), c);
      c = wmma_bf16(aq1, load_bt(Ks + nt * 16 * 64 + 32, 64, lane), c);
      sacc[nt] = c;
    }

    // ---- online softmax (rows r+8*half, cols across lanes of a 16-group) ----
    float alpha[8];
#pragma unroll
    for (int r = 0; r < 8; ++r) {
      float mx = -1e30f;
#pragma unroll
      for (int nt = 0; nt < 4; ++nt) mx = fmaxf(mx, sacc[nt][r]);
      mx = fmaxf(mx, __shfl_xor(mx, 1, 32));
      mx = fmaxf(mx, __shfl_xor(mx, 2, 32));
      mx = fmaxf(mx, __shfl_xor(mx, 4, 32));
      mx = fmaxf(mx, __shfl_xor(mx, 8, 32));
      mx *= scale;
      const float mnew = fmaxf(mrow[r], mx);
      alpha[r] = __expf(mrow[r] - mnew);
      mrow[r] = mnew;
      float rs = 0.0f;
#pragma unroll
      for (int nt = 0; nt < 4; ++nt) {
        const float p = __expf(sacc[nt][r] * scale - mnew);
        sacc[nt][r] = p;
        rs += p;
      }
      rs += __shfl_xor(rs, 1, 32);
      rs += __shfl_xor(rs, 2, 32);
      rs += __shfl_xor(rs, 4, 32);
      rs += __shfl_xor(rs, 8, 32);
      lrow[r] = lrow[r] * alpha[r] + rs;
    }
#pragma unroll
    for (int r = 0; r < 8; ++r) {
      o0[r] *= alpha[r]; o1[r] *= alpha[r]; o2[r] *= alpha[r]; o3[r] *= alpha[r];
    }

    // ---- repack P (C-layout -> A-layout) through per-wave LDS scratch ----
    const int ncol = lane & 15;
#pragma unroll
    for (int nt = 0; nt < 4; ++nt)
#pragma unroll
      for (int r = 0; r < 8; ++r)
        Pw[(r + 8 * half) * 64 + nt * 16 + ncol] = (bf16_t)sacc[nt][r];
    // same-wave LDS RAW: DScnt wait inserted by compiler
    const v16bf ap0 = load_a(Pw, 64, lane);
    const v16bf ap1 = load_a(Pw + 32, 64, lane);

    // ---- O += P @ V : B[k][n] = Vs[n][k] (contiguous in k) ----
    o0 = wmma_bf16(ap0, load_bt(Vs + 0 * 16 * 64, 64, lane), o0);
    o0 = wmma_bf16(ap1, load_bt(Vs + 0 * 16 * 64 + 32, 64, lane), o0);
    o1 = wmma_bf16(ap0, load_bt(Vs + 1 * 16 * 64, 64, lane), o1);
    o1 = wmma_bf16(ap1, load_bt(Vs + 1 * 16 * 64 + 32, 64, lane), o1);
    o2 = wmma_bf16(ap0, load_bt(Vs + 2 * 16 * 64, 64, lane), o2);
    o2 = wmma_bf16(ap1, load_bt(Vs + 2 * 16 * 64 + 32, 64, lane), o2);
    o3 = wmma_bf16(ap0, load_bt(Vs + 3 * 16 * 64, 64, lane), o3);
    o3 = wmma_bf16(ap1, load_bt(Vs + 3 * 16 * 64 + 32, 64, lane), o3);

    __syncthreads();  // protect Ks/Vs before next block's staging
  }

  // ---- epilogue: normalize and scatter to [B][S][H*64] bf16 ----
  const int b_idx = bh >> 4;
  const int h = bh & 15;
  const int ncol = lane & 15;
#pragma unroll
  for (int r = 0; r < 8; ++r) {
    const int s = q0 + r + 8 * half;
    const float inv = 1.0f / lrow[r];
    const long row = ((long)b_idx * SEQ + s) * D_MODEL + h * HEAD_DIM;
    attn[row + 0 * 16 + ncol] = (bf16_t)(o0[r] * inv);
    attn[row + 1 * 16 + ncol] = (bf16_t)(o1[r] * inv);
    attn[row + 2 * 16 + ncol] = (bf16_t)(o2[r] * inv);
    attn[row + 3 * 16 + ncol] = (bf16_t)(o3[r] * inv);
  }
}

// ---------------------------------------------------------------------------
extern "C" void kernel_launch(void* const* d_in, const int* in_sizes, int n_in,
                              void* d_out, int out_size, void* d_ws, size_t ws_size,
                              hipStream_t stream) {
  const float* q  = (const float*)d_in[0];
  const float* k  = (const float*)d_in[1];
  const float* v  = (const float*)d_in[2];
  const float* Wq = (const float*)d_in[3];
  const float* bq = (const float*)d_in[4];
  const float* Wk = (const float*)d_in[5];
  const float* bk = (const float*)d_in[6];
  const float* Wv = (const float*)d_in[7];
  const float* bv = (const float*)d_in[8];
  const float* Wo = (const float*)d_in[9];
  const float* bo = (const float*)d_in[10];
  float* out = (float*)d_out;

  const size_t NE = (size_t)BATCH * SEQ * D_MODEL;  // 8,388,608 elems
  const size_t WN = (size_t)D_MODEL * D_MODEL;      // 1,048,576 elems
  const size_t FULL_ELEMS = 6 * NE + 4 * WN;        // ~104 MB in bf16
  const bool full = ws_size >= FULL_ELEMS * sizeof(bf16_t);

  dim3 blk(128);
  dim3 gproj(MTOT / 32, D_MODEL / 128);  // 256 x 8
  dim3 gattn(BHTOT, SEQ / 64);           // 64 x 32

  if (full) {
    // ws layout (bf16 elems): Xq | Xk | Xv | Wqb Wkb Wvb Wob | Qh | Kh | Vt
    // attn output reuses Xq (dead after Q projection).
    bf16_t* Xq  = (bf16_t*)d_ws;
    bf16_t* Xk  = Xq + NE;
    bf16_t* Xv  = Xk + NE;
    bf16_t* Wqb = Xv + NE;
    bf16_t* Wkb = Wqb + WN;
    bf16_t* Wvb = Wkb + WN;
    bf16_t* Wob = Wvb + WN;
    bf16_t* Qh  = Wob + WN;
    bf16_t* Kh  = Qh + NE;
    bf16_t* Vt  = Kh + NE;
    bf16_t* attn = Xq;

    const int n4x = (int)(NE / 4), n4w = (int)(WN / 4);
    cvt_f32_bf16_kernel<<<(n4x + 255) / 256, 256, 0, stream>>>(q, Xq, n4x);
    cvt_f32_bf16_kernel<<<(n4x + 255) / 256, 256, 0, stream>>>(k, Xk, n4x);
    cvt_f32_bf16_kernel<<<(n4x + 255) / 256, 256, 0, stream>>>(v, Xv, n4x);
    cvt_f32_bf16_kernel<<<(n4w + 255) / 256, 256, 0, stream>>>(Wq, Wqb, n4w);
    cvt_f32_bf16_kernel<<<(n4w + 255) / 256, 256, 0, stream>>>(Wk, Wkb, n4w);
    cvt_f32_bf16_kernel<<<(n4w + 255) / 256, 256, 0, stream>>>(Wv, Wvb, n4w);
    cvt_f32_bf16_kernel<<<(n4w + 255) / 256, 256, 0, stream>>>(Wo, Wob, n4w);

    mha_proj_kernel<bf16_t, bf16_t><<<gproj, blk, 0, stream>>>(Xq, Wqb, bq, Qh, 0);
    mha_proj_kernel<bf16_t, bf16_t><<<gproj, blk, 0, stream>>>(Xk, Wkb, bk, Kh, 0);
    mha_proj_kernel<bf16_t, bf16_t><<<gproj, blk, 0, stream>>>(Xv, Wvb, bv, Vt, 1);
    mha_attn_kernel<<<gattn, blk, 0, stream>>>(Qh, Kh, Vt, attn);
    mha_outproj_kernel<bf16_t, bf16_t><<<gproj, blk, 0, stream>>>(attn, Wob, bo, out);
  } else {
    // Fallback (64 MB ws): convert-on-load inside the GEMMs.
    bf16_t* Qh   = (bf16_t*)d_ws;
    bf16_t* Kh   = Qh + NE;
    bf16_t* Vt   = Kh + NE;
    bf16_t* attn = Vt + NE;

    mha_proj_kernel<float, float><<<gproj, blk, 0, stream>>>(q, Wq, bq, Qh, 0);
    mha_proj_kernel<float, float><<<gproj, blk, 0, stream>>>(k, Wk, bk, Kh, 0);
    mha_proj_kernel<float, float><<<gproj, blk, 0, stream>>>(v, Wv, bv, Vt, 1);
    mha_attn_kernel<<<gattn, blk, 0, stream>>>(Qh, Kh, Vt, attn);
    mha_outproj_kernel<bf16_t, float><<<gproj, blk, 0, stream>>>(attn, Wo, bo, out);
  }
}
